// GLSTM_27367531610437
// MI455X (gfx1250) — matrix operand order repaired and checked by reading, not compiled, not run on hardware
//
#include <hip/hip_runtime.h>
#include <stdint.h>
#include <stddef.h>

#define NNODE 2000
#define HIDN  256
#define EMBN  256
#define SEQL  64
#define NTIME 16
#define NFEAT 64
#define NRELS 2
#define NLAY  3
#define NLBL  16

typedef __attribute__((ext_vector_type(16))) __bf16 v16bf;
typedef __attribute__((ext_vector_type(8)))  float  v8f;

union FragBF { v16bf v; unsigned short u[16]; };

__device__ __forceinline__ unsigned short f2bf(float f) {
  union { float f; unsigned int u; } x; x.f = f;
  unsigned int r = x.u + 0x7FFFu + ((x.u >> 16) & 1u);   // RNE
  return (unsigned short)(r >> 16);
}
__device__ __forceinline__ float sigm(float x) { return 1.f / (1.f + expf(-x)); }

enum { FLAG_ACC = 1, FLAG_BIAS = 2, FLAG_TANH = 4 };

// ---------------------------------------------------------------------------
// WMMA GEMM, register-blocked: each wave computes a 16 x (16*NB) strip,
// reusing one A fragment across NB independent v_wmma_f32_16x16x32_bf16.
// A: MxKpad bf16 row-major.  Bp: [Kpad/32][N][32] bf16 fragment layout:
//   lane<16: N=lane, K(tile)=0..15 ; lane>=16: N=lane-16, K(tile)=16..31
// A fragment (16-bit 16x32): lane<16: M=lane, elems 0..15 -> K0..15
//                            lane>=16: K16..31
// 4 waves per block; strip index is wave-uniform so EXEC stays all-1s for WMMA.
// ---------------------------------------------------------------------------
template <int NB>
__global__ void __launch_bounds__(128)
gemm_bf16(const unsigned short* __restrict__ A, const unsigned short* __restrict__ Bp,
          float* __restrict__ C, const float* __restrict__ bias,
          int M, int N, int Kpad, int flags)
{
  int wave = threadIdx.x >> 5;
  int lane = threadIdx.x & 31;
  int stripsN = N / (16 * NB);
  int stripsTotal = (M >> 4) * stripsN;
  int strip = blockIdx.x * 4 + wave;
  if (strip >= stripsTotal) return;             // wave-uniform exit
  int tm  = strip / stripsN;
  int tn0 = (strip % stripsN) * NB;
  int half = lane >> 4, mr = lane & 15;

  v8f acc[NB];
  if (flags & FLAG_ACC) {
#pragma unroll
    for (int j = 0; j < NB; ++j)
#pragma unroll
      for (int i = 0; i < 8; ++i)
        acc[j][i] = C[(size_t)(tm * 16 + half * 8 + i) * N + (tn0 + j) * 16 + mr];
  } else {
#pragma unroll
    for (int j = 0; j < NB; ++j)
#pragma unroll
      for (int i = 0; i < 8; ++i) acc[j][i] = 0.f;
  }

  const int KT = Kpad >> 5;
  for (int kt = 0; kt < KT; ++kt) {
    FragBF a;
    const unsigned short* Ar = A + (size_t)(tm * 16 + mr) * Kpad + kt * 32 + half * 16;
#pragma unroll
    for (int e = 0; e < 8; ++e) a.u[e] = Ar[e];
#pragma unroll
    for (int e = 0; e < 8; ++e) a.u[8 + e] = Ar[8 + e];

    FragBF b[NB];
#pragma unroll
    for (int j = 0; j < NB; ++j) {
      const unsigned short* Br =
          Bp + ((size_t)kt * N + (tn0 + j) * 16 + mr) * 32 + half * 16;
#pragma unroll
      for (int e = 0; e < 16; ++e) b[j].u[e] = Br[e];
    }
    if (kt + 1 < KT) {                 // near-cache (WGP) prefetch of next K-slab
      __builtin_prefetch(Ar + 32, 0, 3);
      __builtin_prefetch(Bp + ((size_t)(kt + 1) * N + tn0 * 16 + mr) * 32, 0, 3);
    }
#pragma unroll
    for (int j = 0; j < NB; ++j)
      acc[j] = __builtin_amdgcn_wmma_f32_16x16x32_bf16(false, a.v, false, b[j].v,
                                                       (short)0, acc[j], false, false);
  }

#pragma unroll
  for (int j = 0; j < NB; ++j) {
#pragma unroll
    for (int i = 0; i < 8; ++i) {
      int row = tm * 16 + half * 8 + i;
      int col = (tn0 + j) * 16 + mr;
      float v = acc[j][i];
      if (flags & FLAG_BIAS) v += bias[col];
      if (flags & FLAG_TANH) v = tanhf(v);
      C[(size_t)row * N + col] = v;
    }
  }
}

// ---- packing / conversion ------------------------------------------------
// f32 (KxN row-major) -> Bp[kt][n][32] bf16, zero-padded to Kpad
__global__ void pack_B(const float* __restrict__ W, unsigned short* __restrict__ Bp,
                       int K, int N, int Kpad)
{
  int idx = blockIdx.x * blockDim.x + threadIdx.x;
  int total = (Kpad >> 5) * N * 32;
  if (idx >= total) return;
  int kk = idx & 31;
  int rest = idx >> 5;
  int n = rest % N;
  int kt = rest / N;
  int k = kt * 32 + kk;
  Bp[idx] = (k < K) ? f2bf(W[(size_t)k * N + n]) : (unsigned short)0;
}

// f32 (MxK row-major) -> bf16 row-major MxKpad (zero pad). K==Kpad => plain convert.
__global__ void pack_A_pad(const float* __restrict__ S, unsigned short* __restrict__ Ap,
                           int M, int K, int Kpad)
{
  int idx = blockIdx.x * blockDim.x + threadIdx.x;
  if (idx >= M * Kpad) return;
  int m = idx / Kpad, k = idx % Kpad;
  Ap[idx] = (k < K) ? f2bf(S[(size_t)m * K + k]) : (unsigned short)0;
}

__global__ void gather_emb_bf16(const int* __restrict__ text, const float* __restrict__ emb,
                                unsigned short* __restrict__ out, int t)
{
  int idx = blockIdx.x * blockDim.x + threadIdx.x;
  if (idx >= NNODE * EMBN) return;
  int n = idx / EMBN, e = idx % EMBN;
  int tok = text[n * SEQL + t];
  out[idx] = f2bf(emb[(size_t)tok * EMBN + e]);
}

__global__ void pack_comb_A(const float* __restrict__ fv, const float* __restrict__ tv,
                            const int* __restrict__ span, const float* __restrict__ nemb,
                            unsigned short* __restrict__ out)
{
  int idx = blockIdx.x * blockDim.x + threadIdx.x;
  if (idx >= NNODE * 3 * HIDN) return;
  int n = idx / (3 * HIDN), j = idx % (3 * HIDN);
  float v;
  if (j < HIDN)           v = fv[(size_t)n * HIDN + j];
  else if (j < 2 * HIDN)  v = tv[(size_t)n * HIDN + (j - HIDN)];
  else                    v = nemb[(size_t)span[n] * HIDN + (j - 2 * HIDN)];
  out[idx] = f2bf(v);
}

__global__ void zero_f32(float* p, int n)
{ int i = blockIdx.x * blockDim.x + threadIdx.x; if (i < n) p[i] = 0.f; }
__global__ void zero_u16(unsigned short* p, int n)
{ int i = blockIdx.x * blockDim.x + threadIdx.x; if (i < n) p[i] = 0; }

// ---- LSTM cells ----------------------------------------------------------
// gate order i,f,u,o (encoder/feature LSTM); optional mask (keep old if 0)
__global__ void lstm_cell(const float* __restrict__ gates, const int* __restrict__ mask,
                          int t, int seqstride, float* __restrict__ c, float* __restrict__ h,
                          unsigned short* __restrict__ hbf)
{
  int idx = blockIdx.x * blockDim.x + threadIdx.x;
  if (idx >= NNODE * HIDN) return;
  int n = idx / HIDN, j = idx % HIDN;
  const float* gr = gates + (size_t)n * 4 * HIDN;
  float gi = gr[j], gf = gr[HIDN + j], gu = gr[2 * HIDN + j], go = gr[3 * HIDN + j];
  float cn = sigm(gf) * c[idx] + sigm(gi) * tanhf(gu);
  float hn = sigm(go) * tanhf(cn);
  bool upd = true;
  if (mask) upd = mask[n * seqstride + t] > 0;
  float cw = upd ? cn : c[idx];
  float hw = upd ? hn : h[idx];
  c[idx] = cw; h[idx] = hw; hbf[idx] = f2bf(hw);
}

// gate order i,f,o,u (graph layer) with broadcast vector gv (len 4H)
__global__ void graph_cell(const float* __restrict__ gates, const float* __restrict__ gv,
                           float* __restrict__ c, float* __restrict__ h,
                           unsigned short* __restrict__ hbf)
{
  int idx = blockIdx.x * blockDim.x + threadIdx.x;
  if (idx >= NNODE * HIDN) return;
  int n = idx / HIDN, j = idx % HIDN;
  const float* gr = gates + (size_t)n * 4 * HIDN;
  float gi = gr[j] + gv[j];
  float gf = gr[HIDN + j] + gv[HIDN + j];
  float go = gr[2 * HIDN + j] + gv[2 * HIDN + j];
  float gu = gr[3 * HIDN + j] + gv[3 * HIDN + j];
  float cn = sigm(gf) * c[idx] + sigm(gi) * tanhf(gu);
  float hn = sigm(go) * tanhf(cn);
  c[idx] = cn; h[idx] = hn; hbf[idx] = f2bf(hn);
}

__global__ void tanh_to_bf16(const float* __restrict__ in, unsigned short* __restrict__ out, int n)
{
  int i = blockIdx.x * blockDim.x + threadIdx.x;
  if (i < n) out[i] = f2bf(tanhf(in[i]));
}

// ---- attention pool pieces ----------------------------------------------
__global__ void dot_rows(const float* __restrict__ T1, const float* __restrict__ u,
                         float* __restrict__ s, int N, int H)
{
  int n = blockIdx.x * blockDim.x + threadIdx.x;
  if (n >= N) return;
  float acc = 0.f;
  for (int k = 0; k < H; ++k) acc += T1[(size_t)n * H + k] * u[k];
  s[n] = acc;
}

__global__ void __launch_bounds__(1024) softmax1d(float* __restrict__ s, int N)
{
  __shared__ float red[1024];
  int tid = threadIdx.x;
  float m = -1e30f;
  for (int i = tid; i < N; i += 1024) m = fmaxf(m, s[i]);
  red[tid] = m; __syncthreads();
  for (int o = 512; o > 0; o >>= 1) { if (tid < o) red[tid] = fmaxf(red[tid], red[tid + o]); __syncthreads(); }
  m = red[0]; __syncthreads();
  float sum = 0.f;
  for (int i = tid; i < N; i += 1024) sum += expf(s[i] - m);
  red[tid] = sum; __syncthreads();
  for (int o = 512; o > 0; o >>= 1) { if (tid < o) red[tid] += red[tid + o]; __syncthreads(); }
  float inv = 1.f / red[0];
  for (int i = tid; i < N; i += 1024) s[i] = expf(s[i] - m) * inv;
}

__global__ void weighted_pool(const float* __restrict__ alpha, const float* __restrict__ mem,
                              float* __restrict__ out, int N, int H)
{
  int h = blockIdx.x * blockDim.x + threadIdx.x;
  if (h >= H) return;
  float s = 0.f;
  for (int n = 0; n < N; ++n) s += alpha[n] * mem[(size_t)n * H + h];
  out[h] = s;
}

// ---- small GEMVs: out[n] = act( x@W [+ x2@W2] + b ) ----------------------
__global__ void gemv(const float* __restrict__ x, const float* __restrict__ W,
                     const float* __restrict__ x2, const float* __restrict__ W2,
                     const float* __restrict__ b, float* __restrict__ out,
                     int K, int N, int act)
{
  int n = blockIdx.x * blockDim.x + threadIdx.x;
  if (n >= N) return;
  float s = b ? b[n] : 0.f;
  for (int k = 0; k < K; ++k) s += x[k] * W[(size_t)k * N + n];
  if (x2) for (int k = 0; k < K; ++k) s += x2[k] * W2[(size_t)k * N + n];
  if (act == 1) s = sigm(s);
  out[n] = s;
}

// ---- per-column softmax over nodes + global cell/gate update -------------
// fw[n,h] = sigmoid(fwlog[n,h] + gw[h] + gub[h]); softmax over n;
// c_g[h] = f_g[h]*c_g[h] + sum_n c[n,h]*fw[n,h]; g[h] = o_g[h]*tanh(c_g[h])
__global__ void __launch_bounds__(256)
colsm_update(const float* __restrict__ fwlog, const float* __restrict__ gw,
             const float* __restrict__ gub, const float* __restrict__ cmat,
             const float* __restrict__ fgog, float* __restrict__ cg, float* __restrict__ g,
             int N, int H)
{
  int h = blockIdx.x, tid = threadIdx.x;
  __shared__ float r1[256], r2[256];
  float base = gw[h] + gub[h];
  float m = -1e30f;
  for (int n = tid; n < N; n += 256) {
    float v = sigm(fwlog[(size_t)n * H + h] + base);
    m = fmaxf(m, v);
  }
  r1[tid] = m; __syncthreads();
  for (int o = 128; o > 0; o >>= 1) { if (tid < o) r1[tid] = fmaxf(r1[tid], r1[tid + o]); __syncthreads(); }
  m = r1[0]; __syncthreads();
  float se = 0.f, ws = 0.f;
  for (int n = tid; n < N; n += 256) {
    float v = sigm(fwlog[(size_t)n * H + h] + base);
    float e = expf(v - m);
    se += e; ws += e * cmat[(size_t)n * H + h];
  }
  r1[tid] = se; r2[tid] = ws; __syncthreads();
  for (int o = 128; o > 0; o >>= 1) { if (tid < o) { r1[tid] += r1[tid + o]; r2[tid] += r2[tid + o]; } __syncthreads(); }
  if (tid == 0) {
    float s = r2[0] / r1[0];
    float cgn = fgog[h] * cg[h] + s;
    cg[h] = cgn;
    g[h] = fgog[HIDN + h] * tanhf(cgn);
  }
}

// ===========================================================================
static inline int cdiv(int a, int b) { return (a + b - 1) / b; }

extern "C" void kernel_launch(void* const* d_in, const int* in_sizes, int n_in,
                              void* d_out, int out_size, void* d_ws, size_t ws_size,
                              hipStream_t stream)
{
  (void)in_sizes; (void)n_in; (void)out_size; (void)ws_size;

  const int*   span     = (const int*)d_in[0];
  const int*   text     = (const int*)d_in[1];
  const int*   tmask    = (const int*)d_in[2];
  const float* nfeat    = (const float*)d_in[3];
  const float* adj      = (const float*)d_in[4];
  const float* word_emb = (const float*)d_in[5];
  const float* node_emb = (const float*)d_in[6];
  const float* enc_Wi   = (const float*)d_in[7];
  const float* enc_Wh   = (const float*)d_in[8];
  const float* enc_b    = (const float*)d_in[9];
  const float* feat_W   = (const float*)d_in[10];
  const float* feat_b   = (const float*)d_in[11];
  const float* fl_Wi    = (const float*)d_in[12];
  const float* fl_Wh    = (const float*)d_in[13];
  const float* fl_b     = (const float*)d_in[14];
  const float* comb_W   = (const float*)d_in[15];
  const float* comb_b   = (const float*)d_in[16];
  const float* ap_w1    = (const float*)d_in[17];
  const float* ap_b1    = (const float*)d_in[18];
  const float* ap_u     = (const float*)d_in[19];
  const float* s_Wh     = (const float*)d_in[20];
  const float* s_Wn     = (const float*)d_in[21];
  const float* s_U      = (const float*)d_in[22];
  const float* s_V      = (const float*)d_in[23];
  const float* s_Vb     = (const float*)d_in[24];
  const float* rg_W     = (const float*)d_in[25];
  const float* rg_b     = (const float*)d_in[26];
  const float* g_W      = (const float*)d_in[27];
  const float* g_U      = (const float*)d_in[28];
  const float* g_Ub     = (const float*)d_in[29];
  const float* g_w      = (const float*)d_in[30];
  const float* g_u      = (const float*)d_in[31];
  const float* g_ub     = (const float*)d_in[32];
  const float* out_W    = (const float*)d_in[33];
  const float* out_b    = (const float*)d_in[34];

  // ---- workspace bump allocator (deterministic) ----
  char* ws = (char*)d_ws;
  size_t off = 0;
  auto alloc = [&](size_t bytes) -> void* {
    void* p = ws + off;
    off += (bytes + 255) & ~(size_t)255;
    return p;
  };
  const int NH  = NNODE * HIDN;          // 512000
  const int KADJ = 2016;                 // 2000 padded to /32

  // packed bf16 weights
  unsigned short* WiEp   = (unsigned short*)alloc((size_t)8 * 1024 * 32 * 2);
  unsigned short* WhEp   = (unsigned short*)alloc((size_t)8 * 1024 * 32 * 2);
  unsigned short* flWip  = (unsigned short*)alloc((size_t)8 * 1024 * 32 * 2);
  unsigned short* flWhp  = (unsigned short*)alloc((size_t)8 * 1024 * 32 * 2);
  unsigned short* featWp = (unsigned short*)alloc((size_t)2 * 256 * 32 * 2);
  unsigned short* combWp = (unsigned short*)alloc((size_t)24 * 256 * 32 * 2);
  unsigned short* apw1p  = (unsigned short*)alloc((size_t)8 * 256 * 32 * 2);
  unsigned short* sWhp   = (unsigned short*)alloc((size_t)8 * 1024 * 32 * 2);
  unsigned short* sUp    = (unsigned short*)alloc((size_t)8 * 1024 * 32 * 2);
  unsigned short* sWnp   = (unsigned short*)alloc((size_t)8 * 1024 * 32 * 2);
  unsigned short* rgWp0  = (unsigned short*)alloc((size_t)8 * 256 * 32 * 2);
  unsigned short* rgWp1  = (unsigned short*)alloc((size_t)8 * 256 * 32 * 2);
  unsigned short* gup    = (unsigned short*)alloc((size_t)8 * 256 * 32 * 2);
  unsigned short* outWp  = (unsigned short*)alloc((size_t)8 * 16 * 32 * 2);
  unsigned short* adjp   = (unsigned short*)alloc((size_t)NRELS * NNODE * KADJ * 2);
  unsigned short* supBp0 = (unsigned short*)alloc((size_t)(KADJ / 32) * 256 * 32 * 2);
  unsigned short* supBp1 = (unsigned short*)alloc((size_t)(KADJ / 32) * 256 * 32 * 2);

  // bf16 activations
  unsigned short* Aact   = (unsigned short*)alloc((size_t)NNODE * 768 * 2);
  unsigned short* hbf    = (unsigned short*)alloc((size_t)NH * 2);
  unsigned short* nvbf   = (unsigned short*)alloc((size_t)NH * 2);
  unsigned short* hnbf   = (unsigned short*)alloc((size_t)NH * 2);

  // f32 activations
  float* gates   = (float*)alloc((size_t)NNODE * 1024 * 4);
  float* c_s     = (float*)alloc((size_t)NH * 4);
  float* text_v  = (float*)alloc((size_t)NH * 4);
  float* feat_v  = (float*)alloc((size_t)NH * 4);
  float* node_v  = (float*)alloc((size_t)NH * 4);
  float* featx   = (float*)alloc((size_t)NH * 4);
  float* sup     = (float*)alloc((size_t)NH * 4);
  float* acc2    = (float*)alloc((size_t)NH * 4);
  float* t1      = (float*)alloc((size_t)NH * 4);
  float* fwlog   = (float*)alloc((size_t)NH * 4);
  float* hg      = (float*)alloc((size_t)NH * 4);
  float* cgn     = (float*)alloc((size_t)NH * 4);
  float* scores  = (float*)alloc((size_t)NNODE * 4);
  float* gvec    = (float*)alloc(256 * 4);
  float* cgvec   = (float*)alloc(256 * 4);
  float* havg    = (float*)alloc(256 * 4);
  float* gv4h    = (float*)alloc(1024 * 4);
  float* fgog    = (float*)alloc(512 * 4);
  float* gwv     = (float*)alloc(256 * 4);

  auto gemm = [&](const unsigned short* A, const unsigned short* Bp, float* C,
                  const float* bias, int M, int N, int Kpad, int flags) {
    if (N % 64 == 0) {
      int strips = (M / 16) * (N / 64);
      gemm_bf16<4><<<cdiv(strips, 4), 128, 0, stream>>>(A, Bp, C, bias, M, N, Kpad, flags);
    } else {
      int strips = (M / 16) * (N / 16);
      gemm_bf16<1><<<cdiv(strips, 4), 128, 0, stream>>>(A, Bp, C, bias, M, N, Kpad, flags);
    }
  };
  auto packW = [&](const float* W, unsigned short* P, int K, int N, int Kpad) {
    int total = (Kpad / 32) * N * 32;
    pack_B<<<cdiv(total, 256), 256, 0, stream>>>(W, P, K, N, Kpad);
  };

  // ---- pack all weights to bf16 fragment layout ----
  packW(enc_Wi, WiEp, 256, 1024, 256);
  packW(enc_Wh, WhEp, 256, 1024, 256);
  packW(fl_Wi, flWip, 256, 1024, 256);
  packW(fl_Wh, flWhp, 256, 1024, 256);
  packW(feat_W, featWp, 64, 256, 64);
  packW(comb_W, combWp, 768, 256, 768);
  packW(ap_w1, apw1p, 256, 256, 256);
  packW(s_Wh, sWhp, 256, 1024, 256);
  packW(s_U, sUp, 256, 1024, 256);
  packW(s_Wn, sWnp, 256, 1024, 256);
  packW(rg_W, rgWp0, 256, 256, 256);
  packW(rg_W + 256 * 256, rgWp1, 256, 256, 256);
  packW(g_u, gup, 256, 256, 256);
  packW(out_W, outWp, 256, 16, 256);
  for (int r = 0; r < NRELS; ++r)
    pack_A_pad<<<cdiv(NNODE * KADJ, 256), 256, 0, stream>>>(
        adj + (size_t)r * NNODE * NNODE, adjp + (size_t)r * NNODE * KADJ,
        NNODE, NNODE, KADJ);

  // ---- encoder LSTM over text (h lives in text_v) ----
  zero_f32<<<cdiv(NH, 256), 256, 0, stream>>>(text_v, NH);
  zero_f32<<<cdiv(NH, 256), 256, 0, stream>>>(c_s, NH);
  zero_u16<<<cdiv(NH, 256), 256, 0, stream>>>(hbf, NH);
  for (int t = 0; t < SEQL; ++t) {
    gather_emb_bf16<<<cdiv(NNODE * EMBN, 256), 256, 0, stream>>>(text, word_emb, Aact, t);
    gemm(Aact, WiEp, gates, enc_b, NNODE, 1024, 256, FLAG_BIAS);
    gemm(hbf, WhEp, gates, nullptr, NNODE, 1024, 256, FLAG_ACC);
    lstm_cell<<<cdiv(NH, 256), 256, 0, stream>>>(gates, tmask, t, SEQL, c_s, text_v, hbf);
  }

  // ---- feature LSTM (h lives in feat_v) ----
  zero_f32<<<cdiv(NH, 256), 256, 0, stream>>>(feat_v, NH);
  zero_f32<<<cdiv(NH, 256), 256, 0, stream>>>(c_s, NH);
  zero_u16<<<cdiv(NH, 256), 256, 0, stream>>>(hbf, NH);
  for (int t = 0; t < NTIME; ++t) {
    pack_A_pad<<<cdiv(NNODE * NFEAT, 256), 256, 0, stream>>>(
        nfeat + (size_t)t * NNODE * NFEAT, Aact, NNODE, NFEAT, NFEAT);
    gemm(Aact, featWp, featx, feat_b, NNODE, 256, 64, FLAG_BIAS);
    pack_A_pad<<<cdiv(NH, 256), 256, 0, stream>>>(featx, Aact, NNODE, 256, 256);
    gemm(Aact, flWip, gates, fl_b, NNODE, 1024, 256, FLAG_BIAS);
    gemm(hbf, flWhp, gates, nullptr, NNODE, 1024, 256, FLAG_ACC);
    lstm_cell<<<cdiv(NH, 256), 256, 0, stream>>>(gates, nullptr, t, SEQL, c_s, feat_v, hbf);
  }

  // ---- combine -> node_vec ----
  pack_comb_A<<<cdiv(NNODE * 768, 256), 256, 0, stream>>>(feat_v, text_v, span, node_emb, Aact);
  gemm(Aact, combWp, node_v, comb_b, NNODE, 256, 768, FLAG_BIAS);
  pack_A_pad<<<cdiv(NH, 256), 256, 0, stream>>>(node_v, nvbf, NNODE, 256, 256);

  // graph state init: h = c = node_vec
  hipMemcpyAsync(hg, node_v, (size_t)NH * 4, hipMemcpyDeviceToDevice, stream);
  hipMemcpyAsync(cgn, node_v, (size_t)NH * 4, hipMemcpyDeviceToDevice, stream);
  hipMemcpyAsync(hbf, nvbf, (size_t)NH * 2, hipMemcpyDeviceToDevice, stream);

  auto pool = [&](const unsigned short* membf, const float* memf, float* outv) {
    gemm(membf, apw1p, t1, ap_b1, NNODE, 256, 256, FLAG_BIAS | FLAG_TANH);
    dot_rows<<<cdiv(NNODE, 256), 256, 0, stream>>>(t1, ap_u, scores, NNODE, 256);
    softmax1d<<<1, 1024, 0, stream>>>(scores, NNODE);
    weighted_pool<<<1, 256, 0, stream>>>(scores, memf, outv, NNODE, 256);
  };

  // g = attn_pool(node_vec); c_g identical at init (h == c == node_vec)
  pool(nvbf, node_v, gvec);
  hipMemcpyAsync(cgvec, gvec, 256 * 4, hipMemcpyDeviceToDevice, stream);

  // ---- graph layers ----
  for (int l = 0; l < NLAY; ++l) {
    // sup_r = h @ rg_W[r] + rg_b[r]; packed as B for adj GEMM
    gemm(hbf, rgWp0, sup, rg_b, NNODE, 256, 256, FLAG_BIAS);
    packW(sup, supBp0, NNODE, 256, KADJ);
    gemm(hbf, rgWp1, sup, rg_b + 256, NNODE, 256, 256, FLAG_BIAS);
    packW(sup, supBp1, NNODE, 256, KADJ);
    // hn = tanh(adj0@sup0 + adj1@sup1)
    gemm(adjp, supBp0, acc2, nullptr, NNODE, 256, KADJ, 0);
    gemm(adjp + (size_t)NNODE * KADJ, supBp1, acc2, nullptr, NNODE, 256, KADJ, FLAG_ACC);
    tanh_to_bf16<<<cdiv(NH, 256), 256, 0, stream>>>(acc2, hnbf, NH);
    // gates = h@s_Wh + node_vec@s_U + hn@s_Wn + (g@s_V + s_Vb)
    gemv<<<cdiv(1024, 256), 256, 0, stream>>>(gvec, s_V, nullptr, nullptr, s_Vb, gv4h, 256, 1024, 0);
    gemm(hbf, sWhp, gates, nullptr, NNODE, 1024, 256, 0);
    gemm(nvbf, sUp, gates, nullptr, NNODE, 1024, 256, FLAG_ACC);
    gemm(hnbf, sWnp, gates, nullptr, NNODE, 1024, 256, FLAG_ACC);
    graph_cell<<<cdiv(NH, 256), 256, 0, stream>>>(gates, gv4h, cgn, hg, hbf);
    // h_avg = attn_pool(h)
    pool(hbf, hg, havg);
    // f_g,o_g = sigmoid(g@g_W + h_avg@g_U + g_Ub)
    gemv<<<cdiv(512, 256), 256, 0, stream>>>(gvec, g_W, havg, g_U, g_Ub, fgog, 256, 512, 1);
    // gw = g@g_w ; fwlog = h@g_u
    gemv<<<1, 256, 0, stream>>>(gvec, g_w, nullptr, nullptr, nullptr, gwv, 256, 256, 0);
    gemm(hbf, gup, fwlog, nullptr, NNODE, 256, 256, 0);
    colsm_update<<<256, 256, 0, stream>>>(fwlog, gwv, g_ub, cgn, fgog, cgvec, gvec, NNODE, 256);
  }

  // ---- outputs: logits then final h ----
  float* out = (float*)d_out;
  gemm(nvbf, outWp, out, out_b, NNODE, NLBL, 256, FLAG_BIAS);
  hipMemcpyAsync(out + NNODE * NLBL, hg, (size_t)NH * 4, hipMemcpyDeviceToDevice, stream);
}
